// GPSA_28810640621630
// MI455X (gfx1250) — matrix operand order repaired and checked by reading, not compiled, not run on hardware
//
#include <hip/hip_runtime.h>
#include <hip/hip_bf16.h>

typedef __attribute__((ext_vector_type(16))) _Float16     v16h;
typedef __attribute__((ext_vector_type(8)))  float        v8f;
typedef __attribute__((ext_vector_type(4)))  unsigned int v4u;
typedef __attribute__((ext_vector_type(8)))  int          v8i;
typedef __attribute__((ext_vector_type(4)))  int          v4i;

#define BATCH 4
#define SEQ   1024
#define CDIM  768
#define NH    12
#define HDIM  64
#define MTOT  (BATCH * SEQ)   // 4096

// ---------------------------------------------------------------------------
// WMMA helpers (gfx1250, wave32). D = A(16x32,f16) x B(32x16,f16) + C(16x16,f32)
// ---------------------------------------------------------------------------
__device__ __forceinline__ v8f wmma32(v16h a, v16h b, v8f c) {
  return __builtin_amdgcn_wmma_f32_16x16x32_f16(
      /*neg_a=*/false, a, /*neg_b=*/false, b,
      /*c_mod=*/(short)0, c, /*reuse_a=*/false, /*reuse_b=*/false);
}

// A-fragment (16x32 f16). base -> A[row0][k0]; rows `stride` halves apart.
// Lane L<16: halves {K0..15} of row L; lane>=16: {K16..31}.
__device__ __forceinline__ v16h load_a_frag(const _Float16* base, int stride) {
  int lane = threadIdx.x & 31;
  const _Float16* p = base + (lane & 15) * stride + ((lane < 16) ? 0 : 16);
  v16h a;
#pragma unroll
  for (int i = 0; i < 8; ++i) { a[i] = p[i]; a[8 + i] = p[8 + i]; }
  return a;
}

// B-fragment for D += A x W^T: B[kk][n] = W[n][kk]. base -> W[col0][k0];
// lane n reads 16 consecutive K values of W-row n (lanes>=16: K+16).
__device__ __forceinline__ v16h load_bT_frag_f16(const _Float16* base, int stride) {
  int lane = threadIdx.x & 31;
  const _Float16* p = base + (lane & 15) * stride + ((lane < 16) ? 0 : 16);
  v16h b;
#pragma unroll
  for (int i = 0; i < 16; ++i) b[i] = p[i];
  return b;
}

// Non-transposed B fragment from LDS: B[kk][n] = Vt[kk][n], rows `stride` apart.
__device__ __forceinline__ v16h load_b_frag_lds(const _Float16* base, int stride) {
  int lane = threadIdx.x & 31;
  int n    = lane & 15;
  int kof  = (lane < 16) ? 0 : 16;
  v16h b;
#pragma unroll
  for (int i = 0; i < 16; ++i) b[i] = base[(kof + i) * stride + n];
  return b;
}

// Generic shared pointer: addr[31:0] is the LDS byte offset (ISA 10.2).
__device__ __forceinline__ unsigned lds_off(const void* p) {
  return (unsigned)(uintptr_t)p;
}

// ---------------------------------------------------------------------------
// Tensor Data Mover: 2D f16 tile (tile_w x tile_h elements) from global to
// LDS, rows packed contiguously (LDS row stride = tile_w). Descriptor per
// ISA ch.8: D# group0/group1, groups 2-3 zero (<=2D tile). One issue per wave.
// ---------------------------------------------------------------------------
__device__ __forceinline__ void tdm_load_2d_f16(
    unsigned lds_addr, const void* gptr,
    unsigned tile_w, unsigned tile_h, unsigned row_stride_elems) {
  unsigned long long ga = (unsigned long long)(uintptr_t)gptr;
  v4u g0;
  g0[0] = 1u;                                        // count=1, user D#
  g0[1] = lds_addr;                                  // lds_addr[31:0]
  g0[2] = (unsigned)(ga & 0xFFFFFFFFu);              // global_addr[31:0]
  g0[3] = (unsigned)((ga >> 32) & 0x01FFFFFFu) | (2u << 30);  // addr[56:32]|type=2
  v8i g1;
  g1[0] = 0x00010000;                                // data_size=1 (2 bytes)
  g1[1] = (int)((tile_w & 0xFFFFu) << 16);           // tensor_dim0[15:0]
  g1[2] = (int)((tile_w >> 16) | ((tile_h & 0xFFFFu) << 16));  // dim0 hi|dim1 lo
  g1[3] = (int)((tile_h >> 16) | (tile_w << 16));    // dim1 hi | tile_dim0
  g1[4] = (int)tile_h;                               // tile_dim1 (tile_dim2=0)
  g1[5] = (int)row_stride_elems;                     // tensor_dim0_stride[31:0]
  g1[6] = 0;
  g1[7] = 0;
  v4i z4 = {0, 0, 0, 0};
#if __clang_major__ >= 23
  v8i z8 = {0, 0, 0, 0, 0, 0, 0, 0};
  __builtin_amdgcn_tensor_load_to_lds(g0, g1, z4, z4, z8, 0);
#else
  __builtin_amdgcn_tensor_load_to_lds(g0, g1, z4, z4, 0);
#endif
}

// ---------------------------------------------------------------------------
// Kernel 0: bulk f32 -> f16 conversion (bandwidth-bound, one-shot).
// ---------------------------------------------------------------------------
__global__ __launch_bounds__(256) void cvt_f32_to_f16(
    const float* __restrict__ s, _Float16* __restrict__ d, int n) {
  int i = (blockIdx.x * 256 + threadIdx.x) * 4;
  if (i + 3 < n) {
    float4 v = *(const float4*)(s + i);
    d[i + 0] = (_Float16)v.x;
    d[i + 1] = (_Float16)v.y;
    d[i + 2] = (_Float16)v.z;
    d[i + 3] = (_Float16)v.w;
  }
}

// ---------------------------------------------------------------------------
// Kernel 1: D[m,n] = sum_k A[m,k]*W[n,k]; all f16 in, f16 out.
// Block = 4 waves, 64x64 tile. The shared 64x32 A tile is DMA'd to LDS by the
// TDM (double-buffered, s_wait_tensorcnt) -- all 4 waves read it from LDS
// instead of 4x-redundant global loads. One B frag / 4 WMMA per k-step / wave.
// ---------------------------------------------------------------------------
__global__ __launch_bounds__(128) void gemm_ff_f16out(
    const _Float16* __restrict__ A, const _Float16* __restrict__ W,
    _Float16* __restrict__ D, int M, int N, int K) {
  __shared__ _Float16 As[2][64 * 32];    // 2 x 4 KB double buffer
  int tid  = threadIdx.x;
  int wave = tid >> 5;
  int m0   = blockIdx.x * 64;
  int n0   = blockIdx.y * 64 + wave * 16;
  v8f acc[4] = {};
  if (wave == 0)
    tdm_load_2d_f16(lds_off(As[0]), A + (size_t)m0 * K, 32, 64, K);
  int buf = 0;
  for (int k0 = 0; k0 < K; k0 += 32) {
    if (wave == 0) {
      if (k0 + 32 < K) {
        tdm_load_2d_f16(lds_off(As[buf ^ 1]),
                        A + (size_t)m0 * K + k0 + 32, 32, 64, K);
        __builtin_amdgcn_s_wait_tensorcnt(1);
      } else {
        __builtin_amdgcn_s_wait_tensorcnt(0);
      }
    }
    __syncthreads();
    __builtin_prefetch(W + (size_t)n0 * K + k0 + 64, 0, 3);
    v16h b = load_bT_frag_f16(W + (size_t)n0 * K + k0, K);
#pragma unroll
    for (int t = 0; t < 4; ++t) {
      v16h a = load_a_frag(As[buf] + t * 16 * 32, 32);
      acc[t] = wmma32(a, b, acc[t]);
    }
    __syncthreads();
    buf ^= 1;
  }
  int lane = tid & 31;
  int roff = (lane < 16) ? 0 : 8;
  int col  = n0 + (lane & 15);
#pragma unroll
  for (int t = 0; t < 4; ++t)
#pragma unroll
    for (int r = 0; r < 8; ++r)
      D[(size_t)(m0 + 16 * t + roff + r) * N + col] = (_Float16)acc[t][r];
}

// ---------------------------------------------------------------------------
// Kernel 2: gated positional self-attention for one (b, h, 16-row q tile).
// qk: f16 [4096][1536] (q at col h*64, k at 768+h*64); v: f16 [4096][768].
// Full 16x1024 score row in LDS; single-pass softmax; analytic positional
// softmax; sigmoid gate; attn@V with double-buffered async-to-LDS V staging.
// ---------------------------------------------------------------------------
__global__ __launch_bounds__(128) void gpsa_attn(
    const _Float16* __restrict__ qk, const _Float16* __restrict__ v,
    const float* __restrict__ pos_w, const float* __restrict__ pos_b,
    const float* __restrict__ gating, _Float16* __restrict__ heads) {
  __shared__ _Float16 S[16 * SEQ];       // 32 KB: scores, then attn weights
  __shared__ _Float16 Vt[2][32 * HDIM];  // 2 x 4 KB async-staged V tiles

  int tid  = threadIdx.x;
  int wave = tid >> 5;
  int lane = tid & 31;
  int qt   = blockIdx.x & 63;            // N/16 = 64 q tiles
  int bh   = blockIdx.x >> 6;
  int b    = bh / NH, h = bh % NH;
  int i0   = qt * 16;

  const int QKW = 2 * CDIM;
  const _Float16* qbase = qk + (size_t)(b * SEQ + i0) * QKW + h * HDIM;
  const _Float16* kbase = qk + (size_t)(b * SEQ) * QKW + CDIM + h * HDIM;

  // Q fragments (K = 0..31, 32..63), reused across all 64 key tiles
  v16h qa0 = load_a_frag(qbase, QKW);
  v16h qa1 = load_a_frag(qbase + 32, QKW);
  const float scale = 0.125f;            // hd^-0.5

  // ---- content scores: S[16][1024] = (Q K^T) * scale --------------------
  for (int t = 0; t < 16; ++t) {
    int j0 = (wave + 4 * t) * 16;
    v16h kb0 = load_bT_frag_f16(kbase + (size_t)j0 * QKW, QKW);
    v16h kb1 = load_bT_frag_f16(kbase + (size_t)j0 * QKW + 32, QKW);
    v8f c = {};
    c = wmma32(qa0, kb0, c);
    c = wmma32(qa1, kb1, c);
    int roff = (lane < 16) ? 0 : 8;
    int col  = j0 + (lane & 15);
#pragma unroll
    for (int r = 0; r < 8; ++r) S[(roff + r) * SEQ + col] = (_Float16)(c[r] * scale);
  }
  __syncthreads();

  // ---- softmax(content) + analytic softmax(positional) + gate -----------
  {
    int row = tid >> 3;                  // 16 rows x 8 lanes each
    int c0  = (tid & 7) * 128;
    int qi  = i0 + row;
    float coef = pos_w[2 * h] + pos_w[2 * h + 1];
    float pb   = pos_b[h];
    float g    = 1.0f / (1.0f + __expf(-gating[h]));
    float smax = -3.0e38f, pmax = -3.0e38f;
    for (int j = c0; j < c0 + 128; ++j) {
      float s  = (float)S[row * SEQ + j];
      float dj = (float)(j - qi);
      float l  = coef * dj * dj + pb;
      smax = fmaxf(smax, s);
      pmax = fmaxf(pmax, l);
    }
#pragma unroll
    for (int m = 1; m < 8; m <<= 1) {
      smax = fmaxf(smax, __shfl_xor(smax, m, 8));
      pmax = fmaxf(pmax, __shfl_xor(pmax, m, 8));
    }
    float ssum = 0.0f, psum = 0.0f;
    for (int j = c0; j < c0 + 128; ++j) {
      float s  = (float)S[row * SEQ + j];
      float dj = (float)(j - qi);
      float l  = coef * dj * dj + pb;
      ssum += __expf(s - smax);
      psum += __expf(l - pmax);
    }
#pragma unroll
    for (int m = 1; m < 8; m <<= 1) {
      ssum += __shfl_xor(ssum, m, 8);
      psum += __shfl_xor(psum, m, 8);
    }
    float wc = (1.0f - g) / ssum;        // both softmaxes sum to 1: combo is
    float wp = g / psum;                 // already normalized
    for (int j = c0; j < c0 + 128; ++j) {
      float s  = (float)S[row * SEQ + j];
      float dj = (float)(j - qi);
      float l  = coef * dj * dj + pb;
      S[row * SEQ + j] =
          (_Float16)(wc * __expf(s - smax) + wp * __expf(l - pmax));
    }
  }
  __syncthreads();

  // ---- out = attn @ V : (16x1024) x (1024x64), wave w owns 16 d-columns --
  // V tiles (32x64 f16) staged by async-to-LDS copies, double-buffered:
  // issue tile mt+1, then s_wait_asynccnt 2 (in-order completion => tile mt
  // has landed), barrier, WMMA on tile mt.
  int dc = wave * 16;
  v8f acc = {};
  int r  = tid >> 2;                     // 32 rows, 4 threads/row
  int d0 = (tid & 3) * 16;               // 16 halves = 32 B per thread
  const _Float16* vrow = v + (size_t)(b * SEQ + r) * CDIM + h * HDIM + d0;
  {
    unsigned ldst = lds_off(&Vt[0][r * HDIM + d0]);
    asm volatile("global_load_async_to_lds_b128 %0, %1, off"
                 :: "v"(ldst), "v"(vrow) : "memory");
    asm volatile("global_load_async_to_lds_b128 %0, %1, off offset:16"
                 :: "v"(ldst), "v"(vrow) : "memory");
  }
  for (int mt = 0; mt < 32; ++mt) {
    if (mt + 1 < 32) {
      const _Float16* vp = vrow + (size_t)(mt + 1) * 32 * CDIM;
      unsigned ldst = lds_off(&Vt[(mt + 1) & 1][r * HDIM + d0]);
      asm volatile("global_load_async_to_lds_b128 %0, %1, off"
                   :: "v"(ldst), "v"(vp) : "memory");
      asm volatile("global_load_async_to_lds_b128 %0, %1, off offset:16"
                   :: "v"(ldst), "v"(vp) : "memory");
      asm volatile("s_wait_asynccnt 2" ::: "memory");
    } else {
      asm volatile("s_wait_asynccnt 0" ::: "memory");
    }
    __syncthreads();
    v16h a  = load_a_frag(S + mt * 32, SEQ);
    v16h bf = load_b_frag_lds(Vt[mt & 1] + dc, HDIM);
    acc = wmma32(a, bf, acc);
    __syncthreads();
  }
  int roff = (lane < 16) ? 0 : 8;
  int col  = h * HDIM + dc + (lane & 15);
#pragma unroll
  for (int rr = 0; rr < 8; ++rr)
    heads[(size_t)(b * SEQ + i0 + roff + rr) * CDIM + col] = (_Float16)acc[rr];
}

// ---------------------------------------------------------------------------
// Kernel 3: out[m,n] = sum_k H[m,k]*W[n,k] + bias[n]; H,W f16; bias/out f32.
// Same TDM-staged 64x64 tiling as kernel 1.
// ---------------------------------------------------------------------------
__global__ __launch_bounds__(128) void gemm_ff_bias_f32out(
    const _Float16* __restrict__ Hm, const _Float16* __restrict__ W,
    const float* __restrict__ bias, float* __restrict__ D, int M, int N, int K) {
  __shared__ _Float16 As[2][64 * 32];
  int tid  = threadIdx.x;
  int wave = tid >> 5;
  int m0   = blockIdx.x * 64;
  int n0   = blockIdx.y * 64 + wave * 16;
  v8f acc[4] = {};
  if (wave == 0)
    tdm_load_2d_f16(lds_off(As[0]), Hm + (size_t)m0 * K, 32, 64, K);
  int buf = 0;
  for (int k0 = 0; k0 < K; k0 += 32) {
    if (wave == 0) {
      if (k0 + 32 < K) {
        tdm_load_2d_f16(lds_off(As[buf ^ 1]),
                        Hm + (size_t)m0 * K + k0 + 32, 32, 64, K);
        __builtin_amdgcn_s_wait_tensorcnt(1);
      } else {
        __builtin_amdgcn_s_wait_tensorcnt(0);
      }
    }
    __syncthreads();
    v16h b = load_bT_frag_f16(W + (size_t)n0 * K + k0, K);
#pragma unroll
    for (int t = 0; t < 4; ++t) {
      v16h a = load_a_frag(As[buf] + t * 16 * 32, 32);
      acc[t] = wmma32(a, b, acc[t]);
    }
    __syncthreads();
    buf ^= 1;
  }
  int lane = tid & 31;
  int roff = (lane < 16) ? 0 : 8;
  int col  = n0 + (lane & 15);
  float bv = bias[col];
#pragma unroll
  for (int t = 0; t < 4; ++t)
#pragma unroll
    for (int r = 0; r < 8; ++r)
      D[(size_t)(m0 + 16 * t + roff + r) * N + col] = acc[t][r] + bv;
}

// ---------------------------------------------------------------------------
extern "C" void kernel_launch(void* const* d_in, const int* in_sizes, int n_in,
                              void* d_out, int out_size, void* d_ws, size_t ws_size,
                              hipStream_t stream) {
  (void)in_sizes; (void)n_in; (void)out_size; (void)ws_size;
  const float* x      = (const float*)d_in[0];
  const float* qk_w   = (const float*)d_in[1];
  const float* v_w    = (const float*)d_in[2];
  const float* proj_w = (const float*)d_in[3];
  const float* proj_b = (const float*)d_in[4];
  const float* pos_w  = (const float*)d_in[5];
  const float* pos_b  = (const float*)d_in[6];
  const float* gating = (const float*)d_in[7];
  float* out = (float*)d_out;

  // ---- workspace partition (all f16) -------------------------------------
  const size_t n_x   = (size_t)MTOT * CDIM;      // 3,145,728
  const size_t n_qkw = (size_t)2 * CDIM * CDIM;  // 1,179,648
  const size_t n_w   = (size_t)CDIM * CDIM;      //   589,824
  char* ws = (char*)d_ws;
  _Float16* xh    = (_Float16*)ws;                      ws += n_x   * 2;
  _Float16* qkwh  = (_Float16*)ws;                      ws += n_qkw * 2;
  _Float16* vwh   = (_Float16*)ws;                      ws += n_w   * 2;
  _Float16* pwh   = (_Float16*)ws;                      ws += n_w   * 2;
  _Float16* qkbuf = (_Float16*)ws;                      ws += (size_t)MTOT * 1536 * 2;
  _Float16* vbuf  = (_Float16*)ws;                      ws += (size_t)MTOT * CDIM * 2;
  _Float16* hbuf  = (_Float16*)ws;

  dim3 blk(128);
  // one-shot f32 -> f16 conversions (bandwidth-bound)
  cvt_f32_to_f16<<<dim3((unsigned)(n_x   / 1024)), 256, 0, stream>>>(x,      xh,   (int)n_x);
  cvt_f32_to_f16<<<dim3((unsigned)(n_qkw / 1024)), 256, 0, stream>>>(qk_w,   qkwh, (int)n_qkw);
  cvt_f32_to_f16<<<dim3((unsigned)(n_w   / 1024)), 256, 0, stream>>>(v_w,    vwh,  (int)n_w);
  cvt_f32_to_f16<<<dim3((unsigned)(n_w   / 1024)), 256, 0, stream>>>(proj_w, pwh,  (int)n_w);

  gemm_ff_f16out<<<dim3(MTOT / 64, 1536 / 64), blk, 0, stream>>>(
      xh, qkwh, qkbuf, MTOT, 1536, CDIM);
  gemm_ff_f16out<<<dim3(MTOT / 64, CDIM / 64), blk, 0, stream>>>(
      xh, vwh, vbuf, MTOT, CDIM, CDIM);
  gpsa_attn<<<dim3(BATCH * NH * (SEQ / 16)), blk, 0, stream>>>(
      qkbuf, vbuf, pos_w, pos_b, gating, hbuf);
  gemm_ff_bias_f32out<<<dim3(MTOT / 64, CDIM / 64), blk, 0, stream>>>(
      hbuf, pwh, proj_b, out, MTOT, CDIM, CDIM);
}
